// Transformer3DETR_69741678952933
// MI455X (gfx1250) — matrix operand order repaired and checked by reading, not compiled, not run on hardware
//
#include <hip/hip_runtime.h>
#include <hip/hip_bf16.h>

// ---------------- types ----------------
typedef __attribute__((ext_vector_type(16))) _Float16 v16h;
typedef __attribute__((ext_vector_type(8)))  _Float16 v8h;
typedef __attribute__((ext_vector_type(8)))  float    v8f;
typedef __attribute__((ext_vector_type(4)))  unsigned int v4u;
typedef __attribute__((ext_vector_type(8)))  int v8i;
typedef __attribute__((ext_vector_type(4)))  int v4i;

// ---------------- model constants ----------------
#define CH    144
#define NHEAD 8
#define HDIM  18          // 144/8
#define NQ_   100
#define NCLS_ 10
#define NSAMP_ 1024
#define FF_   2048
#define NB    8
#define NPTS  65536
#define NTOK  (NB*NSAMP_)     // 8192
#define NDQ   (NB*NQ_)        // 800

#if __has_builtin(__builtin_amdgcn_tensor_load_to_lds)
#define HAVE_TDM 1
#endif

__device__ inline v8f wmma_f16(v16h a, v16h b, v8f c) {
  return __builtin_amdgcn_wmma_f32_16x16x32_f16(false, a, false, b, (short)0, c, false, false);
}

union Frag16 { v16h v; v8h h[2]; };

// =======================================================================
// 1) per-batch min/max of xyz (for pos-embed normalization)
// =======================================================================
__global__ __launch_bounds__(256) void minmax_kernel(const float* __restrict__ x,
                                                     float* __restrict__ mnmx) {
  __shared__ float sred[256 * 6];
  int b = blockIdx.x, t = threadIdx.x;
  const float* xb = x + (size_t)b * NPTS * 3;
  float mn[3] = {1e30f, 1e30f, 1e30f}, mx[3] = {-1e30f, -1e30f, -1e30f};
  for (int i = t; i < NPTS; i += 256) {
    for (int c = 0; c < 3; ++c) {
      float v = xb[i * 3 + c];
      mn[c] = fminf(mn[c], v);
      mx[c] = fmaxf(mx[c], v);
    }
  }
  for (int c = 0; c < 3; ++c) { sred[t * 6 + c] = mn[c]; sred[t * 6 + 3 + c] = mx[c]; }
  __syncthreads();
  for (int off = 128; off > 0; off >>= 1) {
    if (t < off) {
      for (int c = 0; c < 3; ++c) {
        sred[t * 6 + c]     = fminf(sred[t * 6 + c],     sred[(t + off) * 6 + c]);
        sred[t * 6 + 3 + c] = fmaxf(sred[t * 6 + 3 + c], sred[(t + off) * 6 + 3 + c]);
      }
    }
    __syncthreads();
  }
  if (t == 0) for (int k = 0; k < 6; ++k) mnmx[b * 6 + k] = sred[k];
}

// =======================================================================
// 2) Farthest point sampling: one workgroup per batch, the full 256KB
//    distance array lives in CDNA5's 320KB LDS. 1024 sequential steps.
// =======================================================================
__global__ __launch_bounds__(1024) void fps_kernel(const float* __restrict__ x,
                                                   const int* __restrict__ init_centroid,
                                                   int* __restrict__ idxs) {
  extern __shared__ char smem[];
  float* dist = (float*)smem;                            // NPTS floats (256KB)
  float* redD = (float*)(smem + (size_t)NPTS * 4);       // 1024
  int*   redI = (int*)(smem + (size_t)NPTS * 4 + 4096);  // 1024
  __shared__ int curIdx;
  int b = blockIdx.x, t = threadIdx.x;
  const float* xb = x + (size_t)b * NPTS * 3;
  for (int i = t; i < NPTS; i += 1024) dist[i] = 1e10f;
  if (t == 0) curIdx = init_centroid[b];
  __syncthreads();
  for (int s = 0; s < NSAMP_; ++s) {
    int cur = curIdx;
    if (t == 0) idxs[b * NSAMP_ + s] = cur;
    float cx = xb[cur * 3 + 0], cy = xb[cur * 3 + 1], cz = xb[cur * 3 + 2];
    float bd = -1.0f; int bi = 0;
    for (int i = t; i < NPTS; i += 1024) {
      __builtin_prefetch(xb + (size_t)(i + 1024) * 3, 0, 1);  // global_prefetch
      float dx = xb[i * 3 + 0] - cx;
      float dy = xb[i * 3 + 1] - cy;
      float dz = xb[i * 3 + 2] - cz;
      float d = dx * dx + dy * dy + dz * dz;
      float dm = fminf(dist[i], d);
      dist[i] = dm;
      if (dm > bd) { bd = dm; bi = i; }
    }
    redD[t] = bd; redI[t] = bi;
    __syncthreads();
    for (int off = 512; off > 0; off >>= 1) {
      if (t < off) {
        if (redD[t + off] > redD[t]) { redD[t] = redD[t + off]; redI[t] = redI[t + off]; }
      }
      __syncthreads();
    }
    if (t == 0) curIdx = redI[0];
    __syncthreads();
  }
}

// =======================================================================
// 3) gather sampled points
// =======================================================================
__global__ void gather_kernel(const float* __restrict__ x, const int* __restrict__ idxs,
                              float* __restrict__ smp) {
  int tid = blockIdx.x * blockDim.x + threadIdx.x;
  if (tid >= NTOK) return;
  int b = tid >> 10;
  int src = idxs[tid];
  const float* p = x + ((size_t)b * NPTS + src) * 3;
  smp[tid * 3 + 0] = p[0];
  smp[tid * 3 + 1] = p[1];
  smp[tid * 3 + 2] = p[2];
}

// =======================================================================
// 4) Generic WMMA GEMM: C[M,N] = relu?(A[M,K] @ W[K,N] + bias)
//    Block = 8 waves; each wave owns one 16x16 tile (4 M-tiles x 2 N-tiles).
//    f32 -> f16 staged via LDS, B tile pulled by the Tensor Data Mover.
// =======================================================================
#define G_BM 64
#define G_BN 32
#define G_BK 32
#define A_STR 40   // halves; 80B row stride -> 16B-aligned, conflict-padded
#define B_STR 40

__global__ __launch_bounds__(256) void gemm_wmma(const float* __restrict__ A,
                                                 const float* __restrict__ W,
                                                 const float* __restrict__ bias,
                                                 float* __restrict__ C,
                                                 int M, int K, int Nn, int relu) {
  __shared__ __align__(16) _Float16 As[G_BM * A_STR];
  __shared__ __align__(16) _Float16 Bs[G_BN * B_STR];
#ifdef HAVE_TDM
  __shared__ __align__(16) float Wstage[G_BK * G_BN];
#endif
  int nBase = blockIdx.x * G_BN;
  int mBase = blockIdx.y * G_BM;
  int t = threadIdx.x, lane = t & 31, wave = t >> 5;
  int wm = wave & 3, wn = wave >> 2;
  v8f acc = {};

  // invariant LDS fragment addresses (ISA 7.12.2 layouts)
  int arow = wm * 16 + (lane & 15);
  int acb = (lane >> 4) * 8;
  const v8h* pA0 = (const v8h*)&As[arow * A_STR + acb];
  const v8h* pA1 = (const v8h*)&As[arow * A_STR + acb + 16];
  int brow = wn * 16 + (lane & 15);
  int bkb = (lane >> 4) * 8;
  const v8h* pB0 = (const v8h*)&Bs[brow * B_STR + bkb];
  const v8h* pB1 = (const v8h*)&Bs[brow * B_STR + bkb + 16];

  int kTiles = (K + G_BK - 1) / G_BK;
  for (int kt = 0; kt < kTiles; ++kt) {
    int k0 = kt * G_BK;
    // ---- A tile: [64 rows M][32 cols K] f32->f16
    for (int idx = t; idx < G_BM * G_BK; idx += 256) {
      int r = idx >> 5, c = idx & 31;
      int gm = mBase + r, gk = k0 + c;
      float v = (gm < M && gk < K) ? A[(size_t)gm * K + gk] : 0.0f;
      As[r * A_STR + c] = (_Float16)v;
    }
    // ---- B tile, stored transposed Bs[n][k]
#ifdef HAVE_TDM
    if (wave == 0) {
      // Tensor DMA descriptor (D#) per CDNA5 ISA ch.8: raw fp32 32x32 tile -> LDS
      unsigned long long gaddr =
          (unsigned long long)(uintptr_t)(const void*)(W + (size_t)k0 * Nn + nBase);
      unsigned int ldsOff = (unsigned int)(unsigned long long)(uintptr_t)(void*)Wstage;
      unsigned int td0 = (unsigned int)((Nn - nBase) < G_BN ? (Nn - nBase) : G_BN); // cols
      unsigned int td1 = (unsigned int)((K - k0) < G_BK ? (K - k0) : G_BK);         // rows
      v4u g0;
      g0[0] = 1u;                                           // count=1 valid descriptor
      g0[1] = ldsOff;                                       // lds_addr
      g0[2] = (unsigned int)(gaddr & 0xffffffffull);        // global_addr[31:0]
      g0[3] = (unsigned int)((gaddr >> 32) & 0x1ffffffull) | (2u << 30); // type=image
      v8i g1;
      g1[0] = (int)(2u << 16);                              // data_size = 4 bytes
      g1[1] = (int)((td0 & 0xffffu) << 16);                 // tensor_dim0 lo
      g1[2] = (int)((td0 >> 16) | ((td1 & 0xffffu) << 16)); // dim0 hi | dim1 lo
      g1[3] = (int)((td1 >> 16) | (((unsigned)G_BN) << 16));// dim1 hi | tile_dim0
      g1[4] = (int)(unsigned)G_BK;                          // tile_dim1 | tile_dim2=0
      g1[5] = (int)(unsigned)Nn;                            // tensor_dim0_stride lo
      g1[6] = 0;                                            // stride hi | dim1_stride lo
      g1[7] = 0;
      v4i g2 = {0, 0, 0, 0}, g3 = {0, 0, 0, 0};
#if defined(__clang_major__) && (__clang_major__ >= 23)
      v8i g4 = {0, 0, 0, 0, 0, 0, 0, 0};
      __builtin_amdgcn_tensor_load_to_lds(g0, g1, g2, g3, g4, 0);
#else
      __builtin_amdgcn_tensor_load_to_lds(g0, g1, g2, g3, 0);
#endif
      __builtin_amdgcn_s_wait_tensorcnt(0);
    }
    __syncthreads();
    for (int idx = t; idx < G_BK * G_BN; idx += 256) {
      int r = idx >> 5, c = idx & 31;   // r = K row, c = N col
      Bs[c * B_STR + r] = (_Float16)Wstage[r * G_BN + c];
    }
#else
    for (int idx = t; idx < G_BK * G_BN; idx += 256) {
      int r = idx >> 5, c = idx & 31;
      int gk = k0 + r, gn = nBase + c;
      float v = (gk < K && gn < Nn) ? W[(size_t)gk * Nn + gn] : 0.0f;
      Bs[c * B_STR + r] = (_Float16)v;
    }
#endif
    __syncthreads();
    Frag16 af, bf;
    af.h[0] = *pA0;
    af.h[1] = *pA1;
    bf.h[0] = *pB0;
    bf.h[1] = *pB1;
    acc = wmma_f16(af.v, bf.v, acc);
    __syncthreads();
  }
  // ---- epilogue: bias hoisted (one load per lane), branch-lean stores
  int gn = nBase + wn * 16 + (lane & 15);
  int gmBase = mBase + wm * 16 + 8 * (lane >> 4);
  bool nok = (gn < Nn);
  float bv = nok ? bias[gn] : 0.0f;
#pragma unroll
  for (int r = 0; r < 8; ++r) {
    int gm = gmBase + r;
    if (nok && gm < M) {
      float v = acc[r] + bv;
      v = relu ? fmaxf(v, 0.0f) : v;
      C[(size_t)gm * Nn + gn] = v;
    }
  }
}

// =======================================================================
// 5) Attention: one wave per (b, head, 16-query tile). Scores tile in LDS,
//    QK^T and P*V both through WMMA (HD=18 padded to K=32 / two N-tiles).
// =======================================================================
__global__ __launch_bounds__(32) void attn_kernel(const float* __restrict__ Qp,
                                                  const float* __restrict__ Kp,
                                                  const float* __restrict__ Vp,
                                                  float* __restrict__ O,
                                                  int Sq, int Skv, float scale) {
  extern __shared__ char smem[];
  float* scores = (float*)smem;                               // [16][Skv]
  _Float16* qs = (_Float16*)(smem + (size_t)16 * Skv * 4);    // [16][40]
  _Float16* ks = qs + 16 * 40;                                // [16][40]
  _Float16* pt = ks + 16 * 40;                                // [16][40]
  _Float16* vt = pt + 16 * 40;                                // [32][40]
  int qt = blockIdx.x, h = blockIdx.y, b = blockIdx.z;
  int lane = threadIdx.x;
  int qBase = qt * 16;
  int colBase = h * HDIM;
  const float* Qb = Qp + (size_t)b * Sq * CH;
  const float* Kb = Kp + (size_t)b * Skv * CH;
  const float* Vb = Vp + (size_t)b * Skv * CH;

  // Q tile (f32 -> f16, pad HD 18 -> 32)
  for (int idx = lane; idx < 16 * 32; idx += 32) {
    int r = idx >> 5, c = idx & 31;
    int q = qBase + r;
    float v = (q < Sq && c < HDIM) ? Qb[(size_t)q * CH + colBase + c] : 0.0f;
    qs[r * 40 + c] = (_Float16)v;
  }
  Frag16 aq;
  int arow = (lane & 15);
  int acb = (lane >> 4) * 8;
  aq.h[0] = *(const v8h*)&qs[arow * 40 + acb];
  aq.h[1] = *(const v8h*)&qs[arow * 40 + acb + 16];
  int bkb = (lane >> 4) * 8;

  // ---- scores = scale * Q K^T
  int ktiles = (Skv + 15) / 16;
  for (int kt = 0; kt < ktiles; ++kt) {
    for (int idx = lane; idx < 16 * 32; idx += 32) {
      int r = idx >> 5, c = idx & 31;
      int k = kt * 16 + r;
      float v = (k < Skv && c < HDIM) ? Kb[(size_t)k * CH + colBase + c] : 0.0f;
      ks[r * 40 + c] = (_Float16)v;
    }
    Frag16 bk;
    bk.h[0] = *(const v8h*)&ks[(lane & 15) * 40 + bkb];
    bk.h[1] = *(const v8h*)&ks[(lane & 15) * 40 + bkb + 16];
    v8f sc = {};
    sc = wmma_f16(aq.v, bk.v, sc);
    int kcol = kt * 16 + (lane & 15);
#pragma unroll
    for (int r = 0; r < 8; ++r) {
      int qr = r + 8 * (lane >> 4);
      if (kcol < Skv) scores[qr * Skv + kcol] = sc[r] * scale;
    }
  }
  // ---- softmax over each of 16 rows (wave-cooperative)
  for (int r = 0; r < 16; ++r) {
    float m = -1e30f;
    for (int c = lane; c < Skv; c += 32) m = fmaxf(m, scores[r * Skv + c]);
    for (int off = 16; off > 0; off >>= 1) m = fmaxf(m, __shfl_xor(m, off, 32));
    float ssum = 0.0f;
    for (int c = lane; c < Skv; c += 32) {
      float e = __expf(scores[r * Skv + c] - m);
      scores[r * Skv + c] = e;
      ssum += e;
    }
    for (int off = 16; off > 0; off >>= 1) ssum += __shfl_xor(ssum, off, 32);
    float inv = 1.0f / ssum;
    for (int c = lane; c < Skv; c += 32) scores[r * Skv + c] *= inv;
  }
  // ---- O = P @ V   (Vt tiles: [n=hd][k=token])
  int ctiles = (Skv + 31) / 32;
  v8f o0 = {}, o1 = {};
  for (int ct = 0; ct < ctiles; ++ct) {
    for (int idx = lane; idx < 16 * 32; idx += 32) {
      int r = idx >> 5, c = idx & 31;
      int k = ct * 32 + c;
      pt[r * 40 + c] = (_Float16)((k < Skv) ? scores[r * Skv + k] : 0.0f);
    }
    for (int idx = lane; idx < 32 * 32; idx += 32) {
      int n2 = idx >> 5, c = idx & 31;
      int k = ct * 32 + c;
      float v = (n2 < HDIM && k < Skv) ? Vb[(size_t)k * CH + colBase + n2] : 0.0f;
      vt[n2 * 40 + c] = (_Float16)v;
    }
    Frag16 ap, bv0, bv1;
    ap.h[0] = *(const v8h*)&pt[arow * 40 + acb];
    ap.h[1] = *(const v8h*)&pt[arow * 40 + acb + 16];
    bv0.h[0] = *(const v8h*)&vt[(lane & 15) * 40 + bkb];
    bv0.h[1] = *(const v8h*)&vt[(lane & 15) * 40 + bkb + 16];
    bv1.h[0] = *(const v8h*)&vt[((lane & 15) + 16) * 40 + bkb];
    bv1.h[1] = *(const v8h*)&vt[((lane & 15) + 16) * 40 + bkb + 16];
    o0 = wmma_f16(ap.v, bv0.v, o0);
    o1 = wmma_f16(ap.v, bv1.v, o1);
  }
#pragma unroll
  for (int r = 0; r < 8; ++r) {
    int qr = qBase + r + 8 * (lane >> 4);
    if (qr < Sq) {
      float* Ob = O + ((size_t)b * Sq + qr) * CH + colBase;
      int n0 = (lane & 15), n1 = 16 + (lane & 15);
      Ob[n0] = o0[r];
      if (n1 < HDIM) Ob[n1] = o1[r];
    }
  }
}

// =======================================================================
// 6) LayerNorm(x + r) * g + b  -- one wave per 144-wide row
// =======================================================================
__global__ __launch_bounds__(256) void ln_res_kernel(const float* __restrict__ X,
                                                     const float* __restrict__ R,
                                                     const float* __restrict__ g,
                                                     const float* __restrict__ bta,
                                                     float* __restrict__ Y, int Mrows) {
  int row = blockIdx.x * 8 + (threadIdx.x >> 5);
  int lane = threadIdx.x & 31;
  if (row >= Mrows) return;
  const float* xr = X + (size_t)row * CH;
  const float* rr = R + (size_t)row * CH;
  float vals[5];
  float s = 0.0f;
#pragma unroll
  for (int i = 0; i < 5; ++i) {
    int c = lane + 32 * i;
    float v = 0.0f;
    if (c < CH) v = xr[c] + rr[c];
    vals[i] = v; s += v;
  }
  for (int off = 16; off > 0; off >>= 1) s += __shfl_xor(s, off, 32);
  float mean = s / (float)CH;
  float vs = 0.0f;
#pragma unroll
  for (int i = 0; i < 5; ++i) {
    int c = lane + 32 * i;
    if (c < CH) { float d = vals[i] - mean; vs += d * d; }
  }
  for (int off = 16; off > 0; off >>= 1) vs += __shfl_xor(vs, off, 32);
  float inv = rsqrtf(vs / (float)CH + 1e-5f);
  float* yr = Y + (size_t)row * CH;
#pragma unroll
  for (int i = 0; i < 5; ++i) {
    int c = lane + 32 * i;
    if (c < CH) yr[c] = (vals[i] - mean) * inv * g[c] + bta[c];
  }
}

// =======================================================================
// 7) mem = f + positional sine embedding of sampled xyz
// =======================================================================
__global__ void posembed_add_kernel(const float* __restrict__ f, const float* __restrict__ smp,
                                    const float* __restrict__ mnmx, float* __restrict__ mem) {
  int tid = blockIdx.x * blockDim.x + threadIdx.x;
  if (tid >= NTOK * CH) return;
  int tok = tid / CH, col = tid - tok * CH;
  int b = tok >> 10;
  int c = col / 48, r = col - 48 * c;
  int j = (r < 24) ? r : (r - 24);
  float mn = mnmx[b * 6 + c], mx = mnmx[b * 6 + 3 + c];
  float v = (smp[tok * 3 + c] - mn) / (mx - mn) * 6.283185307179586f;
  float dimt = __powf(10000.0f, (2.0f * (float)(j >> 1)) / 24.0f);
  float p = v / dimt;
  mem[tid] = f[tid] + ((r < 24) ? __sinf(p) : __cosf(p));
}

// =======================================================================
// 8) broadcast learned queries across batch
// =======================================================================
__global__ void bcastq_kernel(const float* __restrict__ queries, float* __restrict__ QT) {
  int tid = blockIdx.x * blockDim.x + threadIdx.x;
  if (tid >= NDQ * CH) return;
  int row = tid / CH, col = tid - row * CH;
  int q = row % NQ_;
  QT[tid] = queries[q * CH + col];
}

// =======================================================================
// 9) scatter head results into d_out (return-dict insertion order)
// =======================================================================
__global__ void write_out_kernel(const float* __restrict__ LOG, const float* __restrict__ CEN,
                                 const float* __restrict__ SIZ, const float* __restrict__ ANG,
                                 float* __restrict__ out) {
  int i = blockIdx.x * blockDim.x + threadIdx.x;
  if (i >= 54400) return;
  float v;
  if (i < 8000)       v = LOG[24000 + i];            // pred_logits (layer 3)
  else if (i < 10400) v = CEN[7200 + (i - 8000)];    // pred center
  else if (i < 12800) v = SIZ[7200 + (i - 10400)];   // pred size
  else if (i < 13600) v = ANG[2400 + (i - 12800)];   // pred angle
  else if (i < 37600) v = LOG[i - 13600];            // aux logits (layers 0..2)
  else if (i < 44800) v = CEN[i - 37600];            // aux center
  else if (i < 52000) v = SIZ[i - 44800];            // aux size
  else                v = ANG[i - 52000];            // aux angle
  out[i] = v;
}

// =======================================================================
// host orchestration
// =======================================================================
extern "C" void kernel_launch(void* const* d_in, const int* in_sizes, int n_in,
                              void* d_out, int out_size, void* d_ws, size_t ws_size,
                              hipStream_t stream) {
  (void)in_sizes; (void)n_in; (void)out_size; (void)ws_size;
  const float* X = (const float*)d_in[0];
  const int* IC  = (const int*)d_in[1];
  auto F = [&](int i) { return (const float*)d_in[i]; };

  // ---- workspace carve ----
  char* wsb = (char*)d_ws;
  size_t off = 0;
  auto alloc = [&](size_t nf) -> float* { float* p = (float*)(wsb + off); off += nf * 4; return p; };
  int* IDXS = (int*)(wsb); off += (size_t)NTOK * 4;
  float* SMP  = alloc((size_t)NTOK * 3);
  float* MNMX = alloc(64);
  float* TA = alloc((size_t)NTOK * CH);
  float* TB = alloc((size_t)NTOK * CH);
  float* QB = alloc((size_t)NTOK * CH);
  float* KB = alloc((size_t)NTOK * CH);
  float* VB = alloc((size_t)NTOK * CH);
  float* OB = alloc((size_t)NTOK * CH);
  float* HID = alloc((size_t)NTOK * FF_);
  float* QT = alloc((size_t)NDQ * CH);
  float* DQ = alloc((size_t)NDQ * CH);
  float* DK = alloc((size_t)NDQ * CH);
  float* DV = alloc((size_t)NDQ * CH);
  float* DO_ = alloc((size_t)NDQ * CH);
  float* DT = alloc((size_t)NDQ * CH);
  float* STK = alloc((size_t)4 * NDQ * CH);
  float* HT  = alloc((size_t)4 * NDQ * CH);
  float* LOG = alloc((size_t)4 * NDQ * NCLS_);
  float* CEN = alloc((size_t)4 * NDQ * 3);
  float* SIZ = alloc((size_t)4 * NDQ * 3);
  float* ANG = alloc((size_t)4 * NDQ * 1);

  auto gemm = [&](const float* A, const float* W, const float* bias, float* C,
                  int M, int K, int Nn, int relu) {
    dim3 g((Nn + G_BN - 1) / G_BN, (M + G_BM - 1) / G_BM);
    gemm_wmma<<<g, 256, 0, stream>>>(A, W, bias, C, M, K, Nn, relu);
  };
  auto ln = [&](const float* Xp, const float* Rp, const float* g, const float* b,
                float* Y, int M) {
    ln_res_kernel<<<(M + 7) / 8, 256, 0, stream>>>(Xp, Rp, g, b, Y, M);
  };
  const float scale = 0.23570226039551584f;  // 1/sqrt(18)
  const size_t attnLdsFix = (size_t)(16 * 40 * 3 + 32 * 40) * 2;

  // ---- 1. FPS + gather + min/max ----
  minmax_kernel<<<NB, 256, 0, stream>>>(X, MNMX);
  size_t fpsShm = (size_t)NPTS * 4 + 4096 + 4096;
  fps_kernel<<<NB, 1024, fpsShm, stream>>>(X, IC, IDXS);
  gather_kernel<<<(NTOK + 255) / 256, 256, 0, stream>>>(X, IDXS, SMP);

  // ---- 2. point MLP ----
  float* F64 = HID;
  float* F128 = HID + (size_t)NTOK * 64;
  gemm(SMP, F(2), F(3), F64, NTOK, 3, 64, 1);
  gemm(F64, F(4), F(5), F128, NTOK, 64, 128, 1);
  gemm(F128, F(6), F(7), TA, NTOK, 128, CH, 0);

  // ---- 3. encoder (4 layers, post-LN) ----
  for (int l = 0; l < 4; ++l) {
    int bI = 9 + 16 * l;
    gemm(TA, F(bI + 0), F(bI + 4), QB, NTOK, CH, CH, 0);
    gemm(TA, F(bI + 1), F(bI + 5), KB, NTOK, CH, CH, 0);
    gemm(TA, F(bI + 2), F(bI + 6), VB, NTOK, CH, CH, 0);
    size_t shm = (size_t)16 * NSAMP_ * 4 + attnLdsFix;
    attn_kernel<<<dim3(NSAMP_ / 16, NHEAD, NB), 32, shm, stream>>>(QB, KB, VB, OB,
                                                                   NSAMP_, NSAMP_, scale);
    gemm(OB, F(bI + 3), F(bI + 7), QB, NTOK, CH, CH, 0);
    ln(TA, QB, F(bI + 8), F(bI + 9), TB, NTOK);
    gemm(TB, F(bI + 12), F(bI + 13), HID, NTOK, CH, FF_, 1);
    gemm(HID, F(bI + 14), F(bI + 15), OB, NTOK, FF_, CH, 0);
    ln(TB, OB, F(bI + 10), F(bI + 11), TA, NTOK);
  }
  posembed_add_kernel<<<(NTOK * CH + 255) / 256, 256, 0, stream>>>(TA, SMP, MNMX, TB); // TB = mem

  // ---- 4. decoder (4 layers) ----
  bcastq_kernel<<<(NDQ * CH + 255) / 256, 256, 0, stream>>>(F(8), QT);
  size_t shmS = (size_t)16 * NQ_ * 4 + attnLdsFix;
  size_t shmC = (size_t)16 * NSAMP_ * 4 + attnLdsFix;
  for (int l = 0; l < 4; ++l) {
    int bI = 73 + 26 * l;
    // self-attention
    gemm(QT, F(bI + 0), F(bI + 4), DQ, NDQ, CH, CH, 0);
    gemm(QT, F(bI + 1), F(bI + 5), DK, NDQ, CH, CH, 0);
    gemm(QT, F(bI + 2), F(bI + 6), DV, NDQ, CH, CH, 0);
    attn_kernel<<<dim3((NQ_ + 15) / 16, NHEAD, NB), 32, shmS, stream>>>(DQ, DK, DV, DO_,
                                                                        NQ_, NQ_, scale);
    gemm(DO_, F(bI + 3), F(bI + 7), DT, NDQ, CH, CH, 0);
    ln(QT, DT, F(bI + 16), F(bI + 17), QT, NDQ);
    // cross-attention against mem
    gemm(QT, F(bI + 8), F(bI + 12), DQ, NDQ, CH, CH, 0);
    gemm(TB, F(bI + 9), F(bI + 13), KB, NTOK, CH, CH, 0);
    gemm(TB, F(bI + 10), F(bI + 14), VB, NTOK, CH, CH, 0);
    attn_kernel<<<dim3((NQ_ + 15) / 16, NHEAD, NB), 32, shmC, stream>>>(DQ, KB, VB, DO_,
                                                                        NQ_, NSAMP_, scale);
    gemm(DO_, F(bI + 11), F(bI + 15), DT, NDQ, CH, CH, 0);
    ln(QT, DT, F(bI + 18), F(bI + 19), QT, NDQ);
    // FFN
    gemm(QT, F(bI + 22), F(bI + 23), HID, NDQ, CH, FF_, 1);
    gemm(HID, F(bI + 24), F(bI + 25), DT, NDQ, FF_, CH, 0);
    ln(QT, DT, F(bI + 20), F(bI + 21), QT, NDQ);
    hipMemcpyAsync(STK + (size_t)l * NDQ * CH, QT, (size_t)NDQ * CH * 4,
                   hipMemcpyDeviceToDevice, stream);
  }

  // ---- 5. heads on stacked [4*800, 144] ----
  int MS = 4 * NDQ;
  gemm(STK, F(177), F(178), LOG, MS, CH, NCLS_, 0);
  gemm(STK, F(179), F(180), HT, MS, CH, CH, 1);
  gemm(HT, F(181), F(182), CEN, MS, CH, 3, 0);
  gemm(STK, F(183), F(184), HT, MS, CH, CH, 1);
  gemm(HT, F(185), F(186), SIZ, MS, CH, 3, 0);
  gemm(STK, F(187), F(188), HT, MS, CH, CH, 1);
  gemm(HT, F(189), F(190), ANG, MS, CH, 1, 0);

  write_out_kernel<<<(54400 + 255) / 256, 256, 0, stream>>>(LOG, CEN, SIZ, ANG, (float*)d_out);
}
